// StateDecoder_74431783239926
// MI455X (gfx1250) — matrix operand order repaired and checked by reading, not compiled, not run on hardware
//
#include <hip/hip_runtime.h>
#include <hip/hip_bf16.h>
#include <math.h>

typedef __bf16 bf16_t;
typedef __attribute__((ext_vector_type(16))) __bf16 v16bf;
typedef __attribute__((ext_vector_type(8)))  float  v8f;
typedef __attribute__((ext_vector_type(4)))  int    v4i;

#define B_SZ    2048
#define L_SZ    128
#define V_SZ    64
#define LD_SZ   128
#define H_SZ    501
#define D0_SZ   257
#define HP      512     // padded hidden
#define NROWS   1503    // 3*H
#define NT      94      // N tiles of 16 covering 1504
#define KT_H    16      // K tiles of 32 covering 512
#define KT_X0   9       // K tiles of 32 covering 288 (layer-0 x input, D0=257)
#define MT      16      // batch rows per block
#define NBLK    (B_SZ / MT)      // 128 blocks
#define GSTRIDE 1504
#define FRAG_SZ (KT_H * 512)     // bf16 elements per staged 16x512 activation tile (8192)

// ---- A-fragment layout (ISA 7.12.2, 16-bit A 16x32, wave32), bijective (m,k)->slot ----
__device__ __forceinline__ int frag_a_index(int m, int k) {
  int kt = k >> 5, kb = k & 31;
  int ln = m + (((kb >> 3) & 1) << 4);          // kb in 8..15 / 24..31 -> lanes 16..31
  int jj = (kb & 7) + ((kb >> 4) << 3);         // kb>=16 -> upper 8 halves
  return kt * 512 + ln * 16 + jj;
}

__device__ __forceinline__ float sigmoidf_(float x) {
  return 1.f / (1.f + __expf(-x));
}

// ---- gfx1250 async global->LDS copy (ASYNCcnt path), with portable fallback ----
typedef __attribute__((address_space(1))) v4i* gv4i_p;   // global v4i*
typedef __attribute__((address_space(3))) v4i* lv4i_p;   // LDS v4i*

__device__ __forceinline__ void async_copy16(const bf16_t* gsrc, bf16_t* lds_dst) {
#if __has_builtin(__builtin_amdgcn_global_load_async_to_lds_b128)
  __builtin_amdgcn_global_load_async_to_lds_b128((gv4i_p)gsrc, (lv4i_p)lds_dst, 0, 0);
#else
  *(int4*)lds_dst = *(const int4*)gsrc;
#endif
}
__device__ __forceinline__ void async_wait_all() {
#if __has_builtin(__builtin_amdgcn_s_wait_asynccnt)
  __builtin_amdgcn_s_wait_asynccnt(0);
#elif __has_builtin(__builtin_amdgcn_global_load_async_to_lds_b128)
  asm volatile("s_wait_asynccnt 0x0" ::: "memory");
#endif
}

// ---------------- init: zero f32 hidden + bf16 hidden fragments, logits[:,0,:] ----------------
__global__ void init_kernel(float* __restrict__ hws, bf16_t* __restrict__ hfrag,
                            float* __restrict__ out) {
  long long i = (long long)blockIdx.x * blockDim.x + threadIdx.x;
  const long long NH = 3LL * B_SZ * HP;              // == 3*NBLK*FRAG_SZ as well
  if (i < NH) hws[i] = 0.f;
  if (i < 3LL * NBLK * FRAG_SZ) hfrag[i] = (bf16_t)0.f;
  if (i < (long long)B_SZ * V_SZ) {
    int b = (int)(i / V_SZ), v = (int)(i % V_SZ);
    out[(long long)b * L_SZ * V_SZ + v] = (v == 1) ? 1.f : 0.f;
  }
}

// ---------------- weight repack: f32 [N,K] row-major -> bf16 B-fragment tiles ----------------
__global__ void pack_kernel(const float* __restrict__ W, bf16_t* __restrict__ dst,
                            int ntiles, int ktiles, int N, int K) {
  int total = ntiles * ktiles * 512;
  int i = blockIdx.x * blockDim.x + threadIdx.x;
  if (i >= total) return;
  int t = i >> 9;
  int e = i & 511;
  int lane = e >> 4, j = e & 15;
  int nt = t / ktiles, kt = t - nt * ktiles;
  int n = nt * 16 + (lane & 15);
  int k = kt * 32 + (j & 7) + ((lane >> 4) << 3) + ((j >> 3) << 4);
  float v = (n < N && k < K) ? W[(long long)n * K + k] : 0.f;
  dst[i] = (bf16_t)v;
}

// ---------------- fused GRU step: 3 layers + logits for a 16-row batch tile ----------------
__global__ __launch_bounds__(256, 1)
void gru_step_kernel(int step,
    const int*   __restrict__ x_inputs,
    const float* __restrict__ emb,
    const float* __restrict__ z,
    const float* __restrict__ y,
    const bf16_t* __restrict__ pw_gi0, const bf16_t* __restrict__ pw_gh0,
    const bf16_t* __restrict__ pw_gi1, const bf16_t* __restrict__ pw_gh1,
    const bf16_t* __restrict__ pw_gi2, const bf16_t* __restrict__ pw_gh2,
    const bf16_t* __restrict__ pw_lin,
    const float* __restrict__ bih0, const float* __restrict__ bhh0,
    const float* __restrict__ bih1, const float* __restrict__ bhh1,
    const float* __restrict__ bih2, const float* __restrict__ bhh2,
    const float* __restrict__ lin_b,
    float* __restrict__ hws,
    bf16_t* __restrict__ hfrag,          // [3][NBLK][FRAG_SZ] bf16 A-fragment staging
    float* __restrict__ out)
{
  extern __shared__ char smem[];
  float*  gi = (float*)smem;                                   // [16][1504] f32
  float*  gh = (float*)(smem + (size_t)MT * GSTRIDE * 4);      // [16][1504] f32
  bf16_t* Ax = (bf16_t*)(smem + 2u * MT * GSTRIDE * 4);        // 16 ktiles * 512 bf16
  bf16_t* Ah = Ax + FRAG_SZ;                                   // 16 ktiles * 512 bf16

  const int tid  = threadIdx.x;
  const int lane = tid & 31;
  const int wid  = tid >> 5;
  const int r0   = blockIdx.x * MT;

  // ---- async-stage layer-0 A_h from previous step's bf16 fragments (16 KB -> LDS)
  {
    const bf16_t* src = hfrag + (size_t)blockIdx.x * FRAG_SZ;  // layer 0
    for (int i = tid; i < FRAG_SZ / 8; i += 256)               // 1024 x b128
      async_copy16(src + i * 8, Ah + i * 8);
  }
  // ---- stage layer-0 A_x = [emb(tok) | z | y | 0-pad] as bf16 fragments (K padded to 288)
  for (int i = tid; i < MT * 288; i += 256) {
    int m = i / 288, k = i - m * 288;
    int b = r0 + m;
    float v = 0.f;
    if (k < LD_SZ) {
      int tok = (step == 1) ? 1 : x_inputs[b * L_SZ + (step - 1)];
      v = emb[tok * LD_SZ + k];
    } else if (k < 2 * LD_SZ) {
      v = z[b * LD_SZ + (k - LD_SZ)];
    } else if (k == 2 * LD_SZ) {
      v = y[b];
    }
    Ax[frag_a_index(m, k)] = (bf16_t)v;
  }
  async_wait_all();
  __syncthreads();

  for (int l = 0; l < 3; ++l) {
    const bf16_t* pwi = (l == 0) ? pw_gi0 : (l == 1) ? pw_gi1 : pw_gi2;
    const bf16_t* pwh = (l == 0) ? pw_gh0 : (l == 1) ? pw_gh1 : pw_gh2;
    const float*  bi  = (l == 0) ? bih0   : (l == 1) ? bih1   : bih2;
    const float*  bh  = (l == 0) ? bhh0   : (l == 1) ? bhh1   : bhh2;
    const int     ktx = (l == 0) ? KT_X0  : KT_H;
    float* hl = hws + (long long)l * B_SZ * HP;

    // ---- gi = x @ wihᵀ : paired N-tiles, one A load feeds two WMMA chains
    for (int nt0 = wid * 2; nt0 < NT; nt0 += 16) {
      v8f acc0 = {0.f, 0.f, 0.f, 0.f, 0.f, 0.f, 0.f, 0.f};
      v8f acc1 = {0.f, 0.f, 0.f, 0.f, 0.f, 0.f, 0.f, 0.f};
      const v16bf* bp0 = (const v16bf*)(pwi + (long long)nt0 * ktx * 512);
      const v16bf* bp1 = (const v16bf*)(pwi + (long long)(nt0 + 1) * ktx * 512);
      if (nt0 + 16 < NT) __builtin_prefetch(pwi + (long long)(nt0 + 16) * ktx * 512, 0, 3);
      for (int kt = 0; kt < ktx; ++kt) {
        v16bf afrag = *(const v16bf*)(Ax + kt * 512 + lane * 16);
        v16bf b0 = bp0[kt * 32 + lane];
        v16bf b1 = bp1[kt * 32 + lane];
        acc0 = __builtin_amdgcn_wmma_f32_16x16x32_bf16(
            false, afrag, false, b0, (short)0, acc0, false, false);
        acc1 = __builtin_amdgcn_wmma_f32_16x16x32_bf16(
            false, afrag, false, b1, (short)0, acc1, false, false);
      }
      int nc = lane & 15;
      int mb = (lane >> 4) << 3;                 // C layout: lanes>=16 hold M=8..15
      int n0 = nt0 * 16 + nc, n1 = n0 + 16;
      if (n0 < NROWS) {
#pragma unroll
        for (int vv = 0; vv < 8; ++vv) gi[(mb + vv) * GSTRIDE + n0] = acc0[vv];
      }
      if (n1 < NROWS) {
#pragma unroll
        for (int vv = 0; vv < 8; ++vv) gi[(mb + vv) * GSTRIDE + n1] = acc1[vv];
      }
    }
    // ---- gh = h @ whhᵀ (K = 512 padded), same pairing
    for (int nt0 = wid * 2; nt0 < NT; nt0 += 16) {
      v8f acc0 = {0.f, 0.f, 0.f, 0.f, 0.f, 0.f, 0.f, 0.f};
      v8f acc1 = {0.f, 0.f, 0.f, 0.f, 0.f, 0.f, 0.f, 0.f};
      const v16bf* bp0 = (const v16bf*)(pwh + (long long)nt0 * KT_H * 512);
      const v16bf* bp1 = (const v16bf*)(pwh + (long long)(nt0 + 1) * KT_H * 512);
      if (nt0 + 16 < NT) __builtin_prefetch(pwh + (long long)(nt0 + 16) * KT_H * 512, 0, 3);
      for (int kt = 0; kt < KT_H; ++kt) {
        v16bf afrag = *(const v16bf*)(Ah + kt * 512 + lane * 16);
        v16bf b0 = bp0[kt * 32 + lane];
        v16bf b1 = bp1[kt * 32 + lane];
        acc0 = __builtin_amdgcn_wmma_f32_16x16x32_bf16(
            false, afrag, false, b0, (short)0, acc0, false, false);
        acc1 = __builtin_amdgcn_wmma_f32_16x16x32_bf16(
            false, afrag, false, b1, (short)0, acc1, false, false);
      }
      int nc = lane & 15;
      int mb = (lane >> 4) << 3;
      int n0 = nt0 * 16 + nc, n1 = n0 + 16;
      if (n0 < NROWS) {
#pragma unroll
        for (int vv = 0; vv < 8; ++vv) gh[(mb + vv) * GSTRIDE + n0] = acc0[vv];
      }
      if (n1 < NROWS) {
#pragma unroll
        for (int vv = 0; vv < 8; ++vv) gh[(mb + vv) * GSTRIDE + n1] = acc1[vv];
      }
    }
    __syncthreads();   // all waves done reading Ah / writing gi,gh

    // ---- overlap: async-stage NEXT layer's A_h (prev-step h_{l+1} fragments) during gate math
    if (l < 2) {
      const bf16_t* src = hfrag + ((size_t)(l + 1) * NBLK + blockIdx.x) * FRAG_SZ;
      for (int i = tid; i < FRAG_SZ / 8; i += 256)
        async_copy16(src + i * 8, Ah + i * 8);
    }

    // ---- zero bf16 pad slots (j = 501..511) of next A_x staging
    for (int i = tid; i < MT * (HP - H_SZ); i += 256) {
      int m = i / (HP - H_SZ);
      int j = H_SZ + (i - m * (HP - H_SZ));
      Ax[frag_a_index(m, j)] = (bf16_t)0.f;
    }
    // ---- GRU gate math (f32); write h (f32 + bf16 fragments for next step) and stage next A_x
    bf16_t* hfragL = hfrag + ((size_t)l * NBLK + blockIdx.x) * FRAG_SZ;
    for (int i = tid; i < MT * H_SZ; i += 256) {
      int m = i / H_SZ, j = i - m * H_SZ;
      const float* gim = gi + m * GSTRIDE;
      const float* ghm = gh + m * GSTRIDE;
      float r  = sigmoidf_(gim[j]        + bi[j]        + ghm[j]        + bh[j]);
      float zg = sigmoidf_(gim[H_SZ + j] + bi[H_SZ + j] + ghm[H_SZ + j] + bh[H_SZ + j]);
      float hn = ghm[2 * H_SZ + j] + bh[2 * H_SZ + j];
      float nn = tanhf(gim[2 * H_SZ + j] + bi[2 * H_SZ + j] + r * hn);
      long long hidx = (long long)(r0 + m) * HP + j;
      float hprev = hl[hidx];
      float hnew  = (1.f - zg) * nn + zg * hprev;
      hl[hidx] = hnew;
      int fs = frag_a_index(m, j);
      bf16_t hb = (bf16_t)hnew;
      hfragL[fs] = hb;                 // next time step's A_h (global, bf16)
      Ax[fs] = hb;                     // next layer's A_x (LDS)
    }
    async_wait_all();
    __syncthreads();
  }

  // ---- logits = h2 @ lin_wᵀ + lin_b   (V=64 -> 4 N tiles, waves 0..3)
  if (wid < 4) {
    int nt = wid;
    v8f acc = {0.f, 0.f, 0.f, 0.f, 0.f, 0.f, 0.f, 0.f};
    const v16bf* bp = (const v16bf*)(pw_lin + (long long)nt * KT_H * 512);
    for (int kt = 0; kt < KT_H; ++kt) {
      v16bf afrag = *(const v16bf*)(Ax + kt * 512 + lane * 16);
      v16bf bfrag = bp[kt * 32 + lane];
      acc = __builtin_amdgcn_wmma_f32_16x16x32_bf16(
          false, afrag, false, bfrag, (short)0, acc, false, false);
    }
    int n0 = nt * 16 + (lane & 15);
    int mb = (lane >> 4) << 3;
    float bb = lin_b[n0];
#pragma unroll
    for (int vv = 0; vv < 8; ++vv) {
      out[(long long)(r0 + mb + vv) * (L_SZ * V_SZ) + (long long)step * V_SZ + n0] =
          acc[vv] + bb;
    }
  }
}

// ---------------- host ----------------
extern "C" void kernel_launch(void* const* d_in, const int* in_sizes, int n_in,
                              void* d_out, int out_size, void* d_ws, size_t ws_size,
                              hipStream_t stream) {
  const float* z        = (const float*)d_in[0];
  const float* y        = (const float*)d_in[1];
  const int*   x_inputs = (const int*)  d_in[2];
  const float* emb      = (const float*)d_in[3];
  const float* lin_w    = (const float*)d_in[4];
  const float* lin_b    = (const float*)d_in[5];
  const float* wih0 = (const float*)d_in[6];
  const float* whh0 = (const float*)d_in[7];
  const float* bih0 = (const float*)d_in[8];
  const float* bhh0 = (const float*)d_in[9];
  const float* wih1 = (const float*)d_in[10];
  const float* whh1 = (const float*)d_in[11];
  const float* bih1 = (const float*)d_in[12];
  const float* bhh1 = (const float*)d_in[13];
  const float* wih2 = (const float*)d_in[14];
  const float* whh2 = (const float*)d_in[15];
  const float* bih2 = (const float*)d_in[16];
  const float* bhh2 = (const float*)d_in[17];
  float* out = (float*)d_out;

  // workspace carve-up (all chunks 256B-aligned)
  char* ws = (char*)d_ws;
  float* hws = (float*)ws;                                 // [3][B][HP] f32
  size_t off = (size_t)3 * B_SZ * HP * sizeof(float);
  bf16_t* hfrag = (bf16_t*)(ws + off);                     // [3][NBLK][FRAG_SZ] bf16
  off += (size_t)3 * NBLK * FRAG_SZ * sizeof(bf16_t);
  const size_t sz_gi0 = (size_t)NT * KT_X0 * 512 * sizeof(bf16_t);
  const size_t sz_ghX = (size_t)NT * KT_H  * 512 * sizeof(bf16_t);
  const size_t sz_lin = (size_t)4  * KT_H  * 512 * sizeof(bf16_t);
  bf16_t* pw_gi0 = (bf16_t*)(ws + off); off += sz_gi0;
  bf16_t* pw_gh0 = (bf16_t*)(ws + off); off += sz_ghX;
  bf16_t* pw_gi1 = (bf16_t*)(ws + off); off += sz_ghX;
  bf16_t* pw_gh1 = (bf16_t*)(ws + off); off += sz_ghX;
  bf16_t* pw_gi2 = (bf16_t*)(ws + off); off += sz_ghX;
  bf16_t* pw_gh2 = (bf16_t*)(ws + off); off += sz_ghX;
  bf16_t* pw_lin = (bf16_t*)(ws + off); off += sz_lin;

  const size_t smem = 2u * MT * GSTRIDE * 4 + 2u * FRAG_SZ * sizeof(bf16_t); // 225280 B
  (void)hipFuncSetAttribute((const void*)gru_step_kernel,
                            hipFuncAttributeMaxDynamicSharedMemorySize, (int)smem);

  {
    long long n = 3LL * B_SZ * HP;
    init_kernel<<<(int)((n + 255) / 256), 256, 0, stream>>>(hws, hfrag, out);
  }
  auto pack = [&](const float* W, bf16_t* dst, int ntiles, int ktiles, int N, int K) {
    int total = ntiles * ktiles * 512;
    pack_kernel<<<(total + 255) / 256, 256, 0, stream>>>(W, dst, ntiles, ktiles, N, K);
  };
  pack(wih0, pw_gi0, NT, KT_X0, NROWS, D0_SZ);
  pack(whh0, pw_gh0, NT, KT_H,  NROWS, H_SZ);
  pack(wih1, pw_gi1, NT, KT_H,  NROWS, H_SZ);
  pack(whh1, pw_gh1, NT, KT_H,  NROWS, H_SZ);
  pack(wih2, pw_gi2, NT, KT_H,  NROWS, H_SZ);
  pack(whh2, pw_gh2, NT, KT_H,  NROWS, H_SZ);
  pack(lin_w, pw_lin, 4, KT_H,  V_SZ,  H_SZ);

  for (int t = 1; t < L_SZ; ++t) {
    gru_step_kernel<<<NBLK, 256, smem, stream>>>(
        t, x_inputs, emb, z, y,
        pw_gi0, pw_gh0, pw_gi1, pw_gh1, pw_gi2, pw_gh2, pw_lin,
        bih0, bhh0, bih1, bhh1, bih2, bhh2, lin_b, hws, hfrag, out);
  }
}
